// LinkPredictor_49460843381232
// MI455X (gfx1250) — matrix operand order repaired and checked by reading, not compiled, not run on hardware
//
#include <hip/hip_runtime.h>

typedef __attribute__((ext_vector_type(16))) _Float16 v16h;
typedef __attribute__((ext_vector_type(8)))  float    v8f;

constexpr int HID = 256;   // hidden
constexpr int H2  = 512;   // 2*hidden
constexpr int IND = 256;   // in_drug
constexpr int INP = 512;   // in_prot
constexpr float BN_EPS = 1e-5f;

__device__ __forceinline__ int laneid() { return threadIdx.x & 31; }
__device__ __forceinline__ int waveid() { return threadIdx.x >> 5; }

__device__ __forceinline__ v8f wmma_f16(v16h a, v16h b, v8f c) {
  // D = A(16x32 f16) * B(32x16 f16) + C(16x16 f32)
  return __builtin_amdgcn_wmma_f32_16x16x32_f16(
      false, a, false, b, (short)0, c, false, false);
}

// A fragment: 16 rows x 32 K, row-major source (LDS), stride in halfs.
// ISA 05_wmma layout: lanes0-15 V0..3:K=0..7, V4..7:K=16..23; lanes16-31 +8.
__device__ __forceinline__ v16h load_a_frag(const _Float16* base, int stride,
                                            int kbase, int lane) {
  const int m = lane & 15, hh = lane >> 4;
  const _Float16* r = base + m * stride + kbase;
  v16h a;
#pragma unroll
  for (int v = 0; v < 8; ++v) {
    const int k0 = (v < 4 ? 2 * v : 16 + 2 * (v - 4)) + 8 * hh;
    a[2 * v]     = r[k0];
    a[2 * v + 1] = r[k0 + 1];
  }
  return a;
}

// B fragment: B[k][n] = W[obase+n][kbase+k], W row-major [O][K] (f16).
// Layout: lane n=L&15; lanes0-15 V0..3:K=0..7, V4..7:K=16..23; lanes16-31 +8.
__device__ __forceinline__ v16h load_b_frag_h(const _Float16* W, int K,
                                              int obase, int kbase, int lane) {
  const int n = lane & 15, hh = lane >> 4;
  const _Float16* r = W + (size_t)(obase + n) * K + kbase + 8 * hh;
  v16h b;
#pragma unroll
  for (int j = 0; j < 8; ++j) { const int kj = j < 4 ? 2 * j : 16 + 2 * (j - 4); b[2 * j] = r[kj]; b[2 * j + 1] = r[kj + 1]; }
  return b;
}

// ---------------------------------------------------------------- utilities
__global__ void zero_f32(float* p, int n) {
  int i = blockIdx.x * blockDim.x + threadIdx.x;
  if (i < n) p[i] = 0.f;
}

__global__ void f32_to_f16(const float* __restrict__ s, _Float16* __restrict__ d, int n) {
  int i = blockIdx.x * blockDim.x + threadIdx.x;
  if (i < n) d[i] = (_Float16)s[i];
}

// ------------------------------------------------ generic C = A @ W^T GEMM
// A [M,K] f32 row-major, Wh [Nc,K] f16 row-major; 64 rows/block, 8 waves.
// Each wave owns an N-tile and accumulates 4 C tiles per B fragment.
__global__ void gemm_nt(const float* __restrict__ A, int M, int K,
                        const _Float16* __restrict__ Wh, int Nc,
                        const float* __restrict__ bias,
                        _Float16* __restrict__ outH, float* __restrict__ outF) {
  extern __shared__ char smem[];
  _Float16* aL = (_Float16*)smem;  // 64 x K
  const int rowBase = blockIdx.x * 64;
  for (int i = threadIdx.x; i < 64 * K; i += blockDim.x) {
    const int m = i / K, k = i - m * K;
    const int gm = rowBase + m;
    aL[i] = (gm < M) ? (_Float16)A[(size_t)gm * K + k] : (_Float16)0.f;
  }
  __syncthreads();
  const int lane = laneid();
  for (int tn = waveid(); tn < Nc / 16; tn += 8) {
    v8f c0 = {}, c1 = {}, c2 = {}, c3 = {};
    __builtin_prefetch(Wh + (size_t)(tn * 16 + (lane & 15)) * K, 0, 1);
    for (int kb = 0; kb < K; kb += 32) {
      v16h b  = load_b_frag_h(Wh, K, tn * 16, kb, lane);
      v16h a0 = load_a_frag(aL + 0 * 16 * K, K, kb, lane);
      v16h a1 = load_a_frag(aL + 1 * 16 * K, K, kb, lane);
      v16h a2 = load_a_frag(aL + 2 * 16 * K, K, kb, lane);
      v16h a3 = load_a_frag(aL + 3 * 16 * K, K, kb, lane);
      c0 = wmma_f16(a0, b, c0);
      c1 = wmma_f16(a1, b, c1);
      c2 = wmma_f16(a2, b, c2);
      c3 = wmma_f16(a3, b, c3);
    }
    const int n = tn * 16 + (lane & 15);
    const float bv = bias ? bias[n] : 0.f;
    const int hh = lane >> 4;
#pragma unroll
    for (int tm = 0; tm < 4; ++tm) {
      v8f c = (tm == 0) ? c0 : (tm == 1) ? c1 : (tm == 2) ? c2 : c3;
#pragma unroll
      for (int r = 0; r < 8; ++r) {
        const int gm = rowBase + tm * 16 + r + 8 * hh;
        if (gm < M) {
          const float v = c[r] + bv;
          if (outH) outH[(size_t)gm * Nc + n] = (_Float16)v;
          if (outF) outF[(size_t)gm * Nc + n] = v;
        }
      }
    }
  }
}

// ---------------------------------------------------------------- stage 1
// gather h_cat, gate GEMM + sigmoid*h, z1 GEMM (+b1), BN1 stats, bilinear dot
__global__ void stage1(const int* __restrict__ edge, int E,
                       const _Float16* __restrict__ Hs, const _Float16* __restrict__ Hd,
                       const _Float16* __restrict__ Vd,
                       const _Float16* __restrict__ Wg, const float* __restrict__ bg,
                       const _Float16* __restrict__ W1h, const float* __restrict__ b1,
                       _Float16* __restrict__ z1, float* __restrict__ bil,
                       float* __restrict__ bnsum, float* __restrict__ bnsq) {
  extern __shared__ char smem[];
  _Float16* hcat = (_Float16*)smem;              // 64 x 512
  _Float16* hg   = hcat + 64 * H2;               // 64 x 512
  float* bnS     = (float*)(hg + 64 * H2);       // 256
  float* bnQ     = bnS + HID;                    // 256
  int*   dId     = (int*)(bnQ + HID);            // 64
  const int eb = blockIdx.x * 64;
  const int valid = min(64, E - eb);

  for (int i = threadIdx.x; i < 2 * HID; i += blockDim.x) bnS[i] = 0.f;

  {  // gather: 4 threads per edge row, 64 halfs of src + 64 of dst each
    const int m = threadIdx.x >> 2, q = threadIdx.x & 3;
    const int e = eb + m;
    const bool ok = e < E;
    int s = 0, d = 0;
    if (ok) { s = edge[e]; d = edge[E + e]; }
    if (q == 0) dId[m] = d;
    const float4* sp = (const float4*)(Hs + (size_t)s * HID) + q * 8;
    const float4* dp = (const float4*)(Hd + (size_t)d * HID) + q * 8;
    float4* o1 = (float4*)(hcat + m * H2) + q * 8;
    float4* o2 = (float4*)(hcat + m * H2 + HID) + q * 8;
    const float4 z4 = {0.f, 0.f, 0.f, 0.f};
#pragma unroll
    for (int i = 0; i < 8; ++i) { o1[i] = ok ? sp[i] : z4; o2[i] = ok ? dp[i] : z4; }
  }
  __syncthreads();

  const int lane = laneid(), wv = waveid();
  // gate GEMM: 64x512 out; each wave owns N-tiles, 4 row tiles per B frag
  for (int tn = wv; tn < 32; tn += 8) {
    v8f c0 = {}, c1 = {}, c2 = {}, c3 = {};
    __builtin_prefetch(Wg + (size_t)(tn * 16 + (lane & 15)) * H2, 0, 1);
    for (int kb = 0; kb < H2; kb += 32) {
      v16h b  = load_b_frag_h(Wg, H2, tn * 16, kb, lane);
      v16h a0 = load_a_frag(hcat + 0 * 16 * H2, H2, kb, lane);
      v16h a1 = load_a_frag(hcat + 1 * 16 * H2, H2, kb, lane);
      v16h a2 = load_a_frag(hcat + 2 * 16 * H2, H2, kb, lane);
      v16h a3 = load_a_frag(hcat + 3 * 16 * H2, H2, kb, lane);
      c0 = wmma_f16(a0, b, c0);
      c1 = wmma_f16(a1, b, c1);
      c2 = wmma_f16(a2, b, c2);
      c3 = wmma_f16(a3, b, c3);
    }
    const int n = tn * 16 + (lane & 15);
    const float bgv = bg[n];
    const int hh = lane >> 4;
#pragma unroll
    for (int tm = 0; tm < 4; ++tm) {
      v8f c = (tm == 0) ? c0 : (tm == 1) ? c1 : (tm == 2) ? c2 : c3;
#pragma unroll
      for (int r = 0; r < 8; ++r) {
        const int m = tm * 16 + r + 8 * hh;
        const float g = 1.f / (1.f + __expf(-(c[r] + bgv)));
        hg[m * H2 + n] = (_Float16)(g * (float)hcat[m * H2 + n]);
      }
    }
  }
  __syncthreads();

  {  // bilinear score: dot(h_src_row, V[dst]) with 4 lanes per edge
    const int m = threadIdx.x >> 2, p = threadIdx.x & 3;
    const int e = eb + m;
    float acc = 0.f;
    if (e < E) {
      const _Float16* s = hcat + m * H2;
      const _Float16* v = Vd + (size_t)dId[m] * HID;
      for (int i = p * 64; i < p * 64 + 64; ++i) acc += (float)s[i] * (float)v[i];
    }
    acc += __shfl_xor(acc, 1, 32);
    acc += __shfl_xor(acc, 2, 32);
    if (p == 0 && e < E) bil[e] = acc;
  }

  // z1 = hg @ W1^T + b1 : 64x256 out; BN1 partial sums
  for (int tn = wv; tn < 16; tn += 8) {
    v8f c0 = {}, c1 = {}, c2 = {}, c3 = {};
    __builtin_prefetch(W1h + (size_t)(tn * 16 + (lane & 15)) * H2, 0, 1);
    for (int kb = 0; kb < H2; kb += 32) {
      v16h b  = load_b_frag_h(W1h, H2, tn * 16, kb, lane);
      v16h a0 = load_a_frag(hg + 0 * 16 * H2, H2, kb, lane);
      v16h a1 = load_a_frag(hg + 1 * 16 * H2, H2, kb, lane);
      v16h a2 = load_a_frag(hg + 2 * 16 * H2, H2, kb, lane);
      v16h a3 = load_a_frag(hg + 3 * 16 * H2, H2, kb, lane);
      c0 = wmma_f16(a0, b, c0);
      c1 = wmma_f16(a1, b, c1);
      c2 = wmma_f16(a2, b, c2);
      c3 = wmma_f16(a3, b, c3);
    }
    const int n = tn * 16 + (lane & 15);
    const float bv = b1[n];
    const int hh = lane >> 4;
    float ps = 0.f, pq = 0.f;
#pragma unroll
    for (int tm = 0; tm < 4; ++tm) {
      v8f c = (tm == 0) ? c0 : (tm == 1) ? c1 : (tm == 2) ? c2 : c3;
#pragma unroll
      for (int r = 0; r < 8; ++r) {
        const int m = tm * 16 + r + 8 * hh;
        if (m < valid) {
          const float v = c[r] + bv;
          z1[(size_t)(eb + m) * HID + n] = (_Float16)v;
          ps += v; pq += v * v;
        }
      }
    }
    atomicAdd(&bnS[n], ps);
    atomicAdd(&bnQ[n], pq);
  }
  __syncthreads();
  for (int i = threadIdx.x; i < HID; i += blockDim.x) {
    atomicAdd(&bnsum[i], bnS[i]);
    atomicAdd(&bnsq[i],  bnQ[i]);
  }
}

// ---------------------------------------------------------------- BN finalize
__global__ void bn_finalize(const float* __restrict__ sum, const float* __restrict__ sq,
                            const float* __restrict__ g, const float* __restrict__ be,
                            float invE, float* __restrict__ a, float* __restrict__ b) {
  const int c = blockIdx.x * blockDim.x + threadIdx.x;
  if (c < HID) {
    const float mean = sum[c] * invE;
    const float var  = sq[c] * invE - mean * mean;
    const float s = g[c] * rsqrtf(var + BN_EPS);
    a[c] = s;
    b[c] = be[c] - mean * s;
  }
}

// ---------------------------------------------------------------- stage 2
// h1 = relu(BN1(z1)); z2 = h1 @ W2^T + b2 (in-place over z buffer); BN2 stats
__global__ void stage2(int E, const _Float16* __restrict__ z1,
                       const float* __restrict__ a1, const float* __restrict__ bb1,
                       const _Float16* __restrict__ W2h, const float* __restrict__ b2,
                       _Float16* __restrict__ z2,
                       float* __restrict__ bnsum, float* __restrict__ bnsq) {
  extern __shared__ char smem[];
  _Float16* h1 = (_Float16*)smem;          // 64 x 256
  float* bnS = (float*)(h1 + 64 * HID);    // 256
  float* bnQ = bnS + HID;                  // 256
  float* aS  = bnQ + HID;                  // 256 (BN1 scale)
  float* bS  = aS + HID;                   // 256 (BN1 shift)
  const int eb = blockIdx.x * 64;
  const int valid = min(64, E - eb);

  for (int i = threadIdx.x; i < 2 * HID; i += blockDim.x) bnS[i] = 0.f;
  for (int i = threadIdx.x; i < HID; i += blockDim.x) { aS[i] = a1[i]; bS[i] = bb1[i]; }
  __syncthreads();
  for (int i = threadIdx.x; i < 64 * HID; i += blockDim.x) {
    const int m = i >> 8, c = i & 255;
    float v = 0.f;
    if (m < valid) {
      v = (float)z1[(size_t)(eb + m) * HID + c];
      v = fmaxf(0.f, v * aS[c] + bS[c]);
    }
    h1[i] = (_Float16)v;
  }
  __syncthreads();

  const int lane = laneid();
  for (int tn = waveid(); tn < 16; tn += 8) {
    v8f c0 = {}, c1 = {}, c2 = {}, c3 = {};
    __builtin_prefetch(W2h + (size_t)(tn * 16 + (lane & 15)) * HID, 0, 1);
    for (int kb = 0; kb < HID; kb += 32) {
      v16h b  = load_b_frag_h(W2h, HID, tn * 16, kb, lane);
      v16h a0 = load_a_frag(h1 + 0 * 16 * HID, HID, kb, lane);
      v16h a1f = load_a_frag(h1 + 1 * 16 * HID, HID, kb, lane);
      v16h a2 = load_a_frag(h1 + 2 * 16 * HID, HID, kb, lane);
      v16h a3 = load_a_frag(h1 + 3 * 16 * HID, HID, kb, lane);
      c0 = wmma_f16(a0, b, c0);
      c1 = wmma_f16(a1f, b, c1);
      c2 = wmma_f16(a2, b, c2);
      c3 = wmma_f16(a3, b, c3);
    }
    const int n = tn * 16 + (lane & 15);
    const float bv = b2[n];
    const int hh = lane >> 4;
    float ps = 0.f, pq = 0.f;
#pragma unroll
    for (int tm = 0; tm < 4; ++tm) {
      v8f c = (tm == 0) ? c0 : (tm == 1) ? c1 : (tm == 2) ? c2 : c3;
#pragma unroll
      for (int r = 0; r < 8; ++r) {
        const int m = tm * 16 + r + 8 * hh;
        if (m < valid) {
          const float v = c[r] + bv;
          z2[(size_t)(eb + m) * HID + n] = (_Float16)v;
          ps += v; pq += v * v;
        }
      }
    }
    atomicAdd(&bnS[n], ps);
    atomicAdd(&bnQ[n], pq);
  }
  __syncthreads();
  for (int i = threadIdx.x; i < HID; i += blockDim.x) {
    atomicAdd(&bnsum[i], bnS[i]);
    atomicAdd(&bnsq[i],  bnQ[i]);
  }
}

// ---------------------------------------------------------------- stage 3
// out[e] = dot(relu(BN2(z2[e])), W3) + b3 + bil[e] + b_bil ; one wave per edge
__global__ void stage3(int E, const _Float16* __restrict__ z2,
                       const float* __restrict__ a2, const float* __restrict__ bb2,
                       const float* __restrict__ W3, const float* __restrict__ b3,
                       const float* __restrict__ bbil, const float* __restrict__ bil,
                       float* __restrict__ out) {
  const int e = blockIdx.x * 8 + waveid();
  if (e >= E) return;
  const int lane = laneid();
  float acc = 0.f;
#pragma unroll
  for (int j = 0; j < 8; ++j) {
    const int c = lane * 8 + j;
    float v = (float)z2[(size_t)e * HID + c];
    v = fmaxf(0.f, v * a2[c] + bb2[c]);
    acc += v * W3[c];
  }
#pragma unroll
  for (int m = 16; m >= 1; m >>= 1) acc += __shfl_xor(acc, m, 32);
  if (lane == 0) out[e] = acc + b3[0] + bil[e] + bbil[0];
}

// ---------------------------------------------------------------- launch
extern "C" void kernel_launch(void* const* d_in, const int* in_sizes, int n_in,
                              void* d_out, int out_size, void* d_ws, size_t ws_size,
                              hipStream_t stream) {
  const float* x_drug = (const float*)d_in[0];
  const float* x_prot = (const float*)d_in[1];
  const int*   edge   = (const int*)d_in[2];
  const float* W_src  = (const float*)d_in[3];
  const float* W_dst  = (const float*)d_in[4];
  const float* W_gate = (const float*)d_in[5];
  const float* b_gate = (const float*)d_in[6];
  const float* W1     = (const float*)d_in[7];
  const float* b1     = (const float*)d_in[8];
  const float* g1     = (const float*)d_in[9];
  const float* be1    = (const float*)d_in[10];
  const float* W2     = (const float*)d_in[11];
  const float* b2     = (const float*)d_in[12];
  const float* g2     = (const float*)d_in[13];
  const float* be2    = (const float*)d_in[14];
  const float* W3     = (const float*)d_in[15];
  const float* b3     = (const float*)d_in[16];
  const float* W_bil  = (const float*)d_in[17];
  const float* b_bil  = (const float*)d_in[18];
  float* out = (float*)d_out;

  const int N = in_sizes[0] / IND;
  const int E = in_sizes[2] / 2;

  char* p = (char*)d_ws;
  auto carve = [&](size_t bytes) -> char* {
    char* r = p; p += (bytes + 255) & ~(size_t)255; return r;
  };
  _Float16* Hs  = (_Float16*)carve((size_t)N * HID * 2);
  _Float16* Hd  = (_Float16*)carve((size_t)N * HID * 2);
  float*    HdF = (float*)   carve((size_t)N * HID * 4);
  _Float16* Vd  = (_Float16*)carve((size_t)N * HID * 2);
  _Float16* WgH = (_Float16*)carve((size_t)H2 * H2 * 2);
  _Float16* W1H = (_Float16*)carve((size_t)HID * H2 * 2);
  _Float16* W2H = (_Float16*)carve((size_t)HID * HID * 2);
  _Float16* WsH = (_Float16*)carve((size_t)HID * IND * 2);
  _Float16* WdH = (_Float16*)carve((size_t)HID * INP * 2);
  _Float16* WbH = (_Float16*)carve((size_t)HID * HID * 2);
  _Float16* Z   = (_Float16*)carve((size_t)E * HID * 2);   // z1, reused as z2
  float*    bil = (float*)   carve((size_t)E * 4);
  float*    bn  = (float*)   carve(4 * HID * 4);           // sum1,sq1,sum2,sq2
  float*    ab  = (float*)   carve(4 * HID * 4);           // a1,b1',a2,b2'

  zero_f32<<<(4 * HID + 255) / 256, 256, 0, stream>>>(bn, 4 * HID);
  f32_to_f16<<<(H2 * H2 + 255) / 256, 256, 0, stream>>>(W_gate, WgH, H2 * H2);
  f32_to_f16<<<(HID * H2 + 255) / 256, 256, 0, stream>>>(W1, W1H, HID * H2);
  f32_to_f16<<<(HID * HID + 255) / 256, 256, 0, stream>>>(W2, W2H, HID * HID);
  f32_to_f16<<<(HID * IND + 255) / 256, 256, 0, stream>>>(W_src, WsH, HID * IND);
  f32_to_f16<<<(HID * INP + 255) / 256, 256, 0, stream>>>(W_dst, WdH, HID * INP);
  f32_to_f16<<<(HID * HID + 255) / 256, 256, 0, stream>>>(W_bil, WbH, HID * HID);

  const int nb = (N + 63) / 64;
  gemm_nt<<<nb, 256, 64 * IND * 2, stream>>>(x_drug, N, IND, WsH, HID,
                                             nullptr, Hs, nullptr);
  gemm_nt<<<nb, 256, 64 * INP * 2, stream>>>(x_prot, N, INP, WdH, HID,
                                             nullptr, Hd, HdF);
  gemm_nt<<<nb, 256, 64 * HID * 2, stream>>>(HdF, N, HID, WbH, HID,
                                             nullptr, Vd, nullptr);

  const int ebks = (E + 63) / 64;
  const size_t s1sh = (size_t)64 * H2 * 2 * 2 + 2 * HID * 4 + 64 * 4;
  stage1<<<ebks, 256, s1sh, stream>>>(edge, E, Hs, Hd, Vd, WgH, b_gate, W1H, b1,
                                      Z, bil, bn, bn + HID);
  bn_finalize<<<1, 256, 0, stream>>>(bn, bn + HID, g1, be1, 1.0f / E, ab, ab + HID);
  const size_t s2sh = (size_t)64 * HID * 2 + 4 * HID * 4;
  stage2<<<ebks, 256, s2sh, stream>>>(E, Z, ab, ab + HID, W2H, b2, Z,
                                      bn + 2 * HID, bn + 3 * HID);
  bn_finalize<<<1, 256, 0, stream>>>(bn + 2 * HID, bn + 3 * HID, g2, be2,
                                     1.0f / E, ab + 2 * HID, ab + 3 * HID);
  stage3<<<(E + 7) / 8, 256, 0, stream>>>(E, Z, ab + 2 * HID, ab + 3 * HID,
                                          W3, b3, b_bil, bil, out);
}
